// HierarchicalEncoder3_37357625541113
// MI455X (gfx1250) — compile-verified, matrix operand order checked by reading, not run on hardware
//
#include <hip/hip_runtime.h>
#include <hip/hip_bf16.h>
#include <math.h>

// Problem constants
#define Bb   8
#define Uu   32
#define Ll   64
#define Hh   512
#define Ss   2048          // U*L
#define Nn   256           // B*U
#define Ee   128
#define Rr   4
#define G4H  2048          // 4*H
#define KK   1024          // H (x) + H (h) concatenated

typedef _Float16 v8h  __attribute__((ext_vector_type(8)));
typedef _Float16 v16h __attribute__((ext_vector_type(16)));
typedef float    v8f  __attribute__((ext_vector_type(8)));

// d_out float offsets (return-order flat):
// enc_h(1,B,H) | enc_c | memory_bank(S,B,H) | lengths(B) | mbu(S,B,H) | lengths(B) | hier(B,S)
#define OFF_ENC_H  0
#define OFF_ENC_C  4096
#define OFF_MB     8192
#define OFF_LEN1   (OFF_MB + Ss*Bb*Hh)            // 8396800
#define OFF_MBU    (OFF_LEN1 + Bb)                // 8396808
#define OFF_LEN2   (OFF_MBU + Ss*Bb*Hh)           // 16785416
#define OFF_HIER   (OFF_LEN2 + Bb)                // 16785424

__device__ __forceinline__ v16h cat8(v8h lo, v8h hi) {
  return __builtin_shufflevector(lo, hi, 0,1,2,3,4,5,6,7,8,9,10,11,12,13,14,15);
}
__device__ __forceinline__ float sigf(float x) { return 1.0f / (1.0f + __expf(-x)); }

// ---------------- prep: X[n][t][h] = f16(emb[word] + spk_tab[spk]) ----------------
__global__ __launch_bounds__(256) void prep_x_kernel(
    const int* __restrict__ src, const int* __restrict__ speaker,
    const float* __restrict__ emb, const float* __restrict__ spk_tab,
    _Float16* __restrict__ X) {
  const int row = blockIdx.x;            // n*64 + t, 16384 rows
  const int n = row >> 6, t = row & 63;
  const int b = n >> 5, u = n & 31;
  const int s = u * Ll + t;
  const int w  = src[s * Bb + b];        // src (S,B,1)
  const int sp = speaker[b * Ss + s];    // speaker (B,S)
  const float* e  = emb    + (size_t)w  * Hh;
  const float* sv = spk_tab + (size_t)sp * Hh;
  _Float16* xr = X + (size_t)row * Hh;
  for (int h = threadIdx.x; h < Hh; h += 256)
    xr[h] = (_Float16)(e[h] + sv[h]);
}

// ---------------- prep: WT[c][k] = f16( k<512 ? W_ih[k][c] : W_hh[k-512][c] ) ----------------
__global__ __launch_bounds__(256) void prep_wt_kernel(
    const float* __restrict__ W_ih, const float* __restrict__ W_hh,
    _Float16* __restrict__ WT) {
  const int c = blockIdx.x;              // 0..2047 (gate column)
  for (int k = threadIdx.x; k < KK; k += 256) {
    float v = (k < Hh) ? W_ih[(size_t)k * G4H + c] : W_hh[(size_t)(k - Hh) * G4H + c];
    WT[(size_t)c * KK + k] = (_Float16)v;
  }
}

// ---------------- LSTM: 16 WGs x 16 sequences, 16 waves/WG, WMMA f16->f32 ----------------
// Each wave owns 32 hidden units (2 col-tiles) x 4 gates = 8 accumulator tiles.
__global__ __launch_bounds__(512) void lstm_kernel(
    const _Float16* __restrict__ X, const _Float16* __restrict__ WT,
    const float* __restrict__ b_lstm,
    float* __restrict__ memory_bank,   // d_out + OFF_MB
    float* __restrict__ utt_h) {
  __shared__ _Float16 hbuf[2][16][Hh];  // 32 KB, double-buffered h (f16)

  const int tid  = threadIdx.x;
  const int wave = tid >> 5;            // 0..15
  const int lane = tid & 31;
  const int ln16 = lane & 15;
  const int hi8  = (lane >> 4) & 1;
  const int wg   = blockIdx.x;          // 0..15
  const int wcol0 = wave * 32;          // this wave's 32 hidden units

  // zero initial h
  for (int i = tid; i < 16 * Hh; i += 512) (&hbuf[0][0][0])[i] = (_Float16)0.0f;
  __syncthreads();

  // per-lane bias for each (gate, col-tile); constant across steps
  float bias[4][2];
#pragma unroll
  for (int gi = 0; gi < 4; ++gi)
#pragma unroll
    for (int ct = 0; ct < 2; ++ct)
      bias[gi][ct] = b_lstm[gi * Hh + wcol0 + ct * 16 + ln16];

  v8f cst[2] = {};                       // cell state, lives in VGPRs all 64 steps

  const int m = ln16;                    // this lane's A-matrix row
  const _Float16* Xrow = X + (size_t)(wg * 16 + m) * Ll * Hh;

  // per-tile weight row base pointers (col = gi*H + wcol0 + ct*16 + ln16)
  const _Float16* wbase[8];
#pragma unroll
  for (int gi = 0; gi < 4; ++gi)
#pragma unroll
    for (int ct = 0; ct < 2; ++ct)
      wbase[gi * 2 + ct] = WT + (size_t)(gi * Hh + wcol0 + ct * 16 + ln16) * KK;

  for (int t = 0; t < Ll; ++t) {
    // Block LICM: weight-fragment loads are t-invariant; without this the
    // compiler hoists 8KB/lane of B-fragments out of the loop and spills
    // them to scratch, serializing every WMMA on a scratch reload.
    asm volatile("" ::: "memory");

    const int rb = t & 1, wb = rb ^ 1;
    v8f acc[8] = {};                     // [gi*2+ct]

    // K = [0,512): A from global X
    for (int ki = 0; ki < 16; ++ki) {
      const int koff = ki * 32 + hi8 * 8;
      const _Float16* pa = Xrow + t * Hh + koff;
      v16h afrag = cat8(*(const v8h*)pa, *(const v8h*)(pa + 16));
      v16h bf[8];
#pragma unroll
      for (int ti = 0; ti < 8; ++ti) {
        const _Float16* q = wbase[ti] + koff;
        bf[ti] = cat8(*(const v8h*)q, *(const v8h*)(q + 16));
      }
#pragma unroll
      for (int ti = 0; ti < 8; ++ti)
        acc[ti] = __builtin_amdgcn_wmma_f32_16x16x32_f16(
            false, afrag, false, bf[ti], (short)0, acc[ti], false, false);
    }
    // K = [512,1024): A from LDS h
    for (int ki = 0; ki < 16; ++ki) {
      const int koff = ki * 32 + hi8 * 8;
      const _Float16* pa = &hbuf[rb][m][koff];
      v16h afrag = cat8(*(const v8h*)pa, *(const v8h*)(pa + 16));
      v16h bf[8];
#pragma unroll
      for (int ti = 0; ti < 8; ++ti) {
        const _Float16* q = wbase[ti] + Hh + koff;
        bf[ti] = cat8(*(const v8h*)q, *(const v8h*)(q + 16));
      }
#pragma unroll
      for (int ti = 0; ti < 8; ++ti)
        acc[ti] = __builtin_amdgcn_wmma_f32_16x16x32_f16(
            false, afrag, false, bf[ti], (short)0, acc[ti], false, false);
    }

    // LSTM cell update — i/f/g/o tiles share the C/D lane layout, pure elementwise
#pragma unroll
    for (int ct = 0; ct < 2; ++ct) {
      const int j = wcol0 + ct * 16 + ln16;           // hidden unit
#pragma unroll
      for (int r = 0; r < 8; ++r) {
        const int row = r + hi8 * 8;                  // which sequence in this WG
        const float iv = sigf(acc[0 * 2 + ct][r] + bias[0][ct]);
        const float fv = sigf(acc[1 * 2 + ct][r] + bias[1][ct]);
        const float gv = tanhf(acc[2 * 2 + ct][r] + bias[2][ct]);
        const float ov = sigf(acc[3 * 2 + ct][r] + bias[3][ct]);
        const float c  = fv * cst[ct][r] + iv * gv;
        cst[ct][r] = c;
        const float h = ov * tanhf(c);
        hbuf[wb][row][j] = (_Float16)h;
        const int seq = wg * 16 + row;
        const int bb = seq >> 5, uu = seq & 31;
        memory_bank[((size_t)(uu * Ll + t) * Bb + bb) * Hh + j] = h;
        if (t == Ll - 1) utt_h[(size_t)seq * Hh + j] = h;
      }
    }
    __syncthreads();
  }
}

// ---------------- GNN: nodes = [utt_h, mean] ----------------
__global__ __launch_bounds__(256) void gnn_nodes_kernel(
    const float* __restrict__ utt_h, float* __restrict__ nodes) {
  const int idx = blockIdx.x * 256 + threadIdx.x;     // B*H = 4096
  const int b = idx >> 9, h = idx & 511;
  float s = 0.0f;
  for (int u = 0; u < Uu; ++u) {
    const float v = utt_h[((size_t)(b * Uu + u)) * Hh + h];
    nodes[((size_t)b * (Uu + 1) + u) * Hh + h] = v;
    s += v;
  }
  nodes[((size_t)b * (Uu + 1) + Uu) * Hh + h] = s * (1.0f / (float)Uu);
}

// ---------------- GNN: msg[e] = nodes[b, src_e] @ W_rel[rel_e] ----------------
__global__ __launch_bounds__(256) void gnn_msg_kernel(
    const float* __restrict__ nodes, const int* __restrict__ edge_src,
    const int* __restrict__ rels, const float* __restrict__ W_rel,
    float* __restrict__ msg) {
  __shared__ float xs[Hh];
  const int e = blockIdx.x;                           // B*E = 1024
  const int b = e >> 7, ei = e & 127;
  const int sn = edge_src[b * Ee + ei];
  const int r  = rels[b * Ee + ei];
  const float* xrow = nodes + ((size_t)b * (Uu + 1) + sn) * Hh;
  for (int k = threadIdx.x; k < Hh; k += 256) xs[k] = xrow[k];
  __syncthreads();
  const float* Wr = W_rel + (size_t)r * Hh * Hh;
  for (int j = threadIdx.x; j < Hh; j += 256) {
    float acc = 0.0f;
    for (int k = 0; k < Hh; ++k) acc += xs[k] * Wr[(size_t)k * Hh + j];
    msg[(size_t)e * Hh + j] = acc;
  }
}

// ---------------- GNN: agg[b,d] = segment_sum over edges ----------------
__global__ __launch_bounds__(256) void gnn_agg_kernel(
    const float* __restrict__ msg, const int* __restrict__ edge_dst,
    float* __restrict__ agg) {
  __shared__ int dst[Ee];
  const int node = blockIdx.x;                        // B*(U+1) = 264
  const int b = node / (Uu + 1), d = node % (Uu + 1);
  for (int e = threadIdx.x; e < Ee; e += 256) dst[e] = edge_dst[b * Ee + e];
  __syncthreads();
  for (int j = threadIdx.x; j < Hh; j += 256) {
    float s = 0.0f;
    for (int e = 0; e < Ee; ++e)
      if (dst[e] == d) s += msg[((size_t)b * Ee + e) * Hh + j];
    agg[(size_t)node * Hh + j] = s;
  }
}

// ---------------- GNN: gated residual update ----------------
__global__ __launch_bounds__(256) void gnn_update_kernel(
    const float* __restrict__ nodes, const float* __restrict__ agg,
    const float* __restrict__ W_self, const float* __restrict__ b_gnn,
    const float* __restrict__ Wg1, const float* __restrict__ Wg2,
    float* __restrict__ new_nodes) {
  __shared__ float nd[Hh];
  __shared__ float ag[Hh];
  const int node = blockIdx.x;                        // 264
  for (int k = threadIdx.x; k < Hh; k += 256) {
    nd[k] = nodes[(size_t)node * Hh + k];
    ag[k] = agg[(size_t)node * Hh + k];
  }
  __syncthreads();
  for (int j = threadIdx.x; j < Hh; j += 256) {
    float s_self = 0.0f, s1 = 0.0f, s2 = 0.0f;
    for (int k = 0; k < Hh; ++k) {
      const float n = nd[k], a = ag[k];
      s_self += n * W_self[(size_t)k * Hh + j];
      s1     += n * Wg1[(size_t)k * Hh + j];
      s2     += a * Wg2[(size_t)k * Hh + j];
    }
    float out = s_self + ag[j] + b_gnn[j];
    out = out > 0.0f ? out : 0.0f;
    const float gate = sigf(s1 + s2);
    new_nodes[(size_t)node * Hh + j] = gate * out + (1.0f - gate) * nd[j];
  }
}

// ---------------- outputs: enc_h/enc_c/lengths/hier ----------------
__global__ __launch_bounds__(256) void finalize_small_kernel(
    const float* __restrict__ new_nodes, float* __restrict__ out) {
  const int tid = blockIdx.x * 256 + threadIdx.x;     // 16384 = B*S
  out[OFF_HIER + tid] = (float)Ll;                    // hier_matrix
  if (tid < Bb * Hh) {
    const int b = tid >> 9, h = tid & 511;
    const float v = new_nodes[((size_t)b * (Uu + 1) + Uu) * Hh + h];  // global node
    out[OFF_ENC_H + tid] = v;
    out[OFF_ENC_C + tid] = v;
  }
  if (tid < Bb) {
    out[OFF_LEN1 + tid] = (float)Ss;
    out[OFF_LEN2 + tid] = (float)Ss;
  }
}

// ---------------- outputs: mbu broadcast (S,B,H) ----------------
__global__ __launch_bounds__(256) void mbu_kernel(
    const float* __restrict__ new_nodes, float* __restrict__ mbu) {
  const size_t o = (size_t)blockIdx.x * 256 + threadIdx.x;   // < S*B*H
  const int h  = (int)(o & 511);
  const int bb = (int)((o >> 9) & 7);
  const int s  = (int)(o >> 12);
  const int u  = s >> 6;
  mbu[o] = new_nodes[((size_t)bb * (Uu + 1) + u) * Hh + h];
}

extern "C" void kernel_launch(void* const* d_in, const int* in_sizes, int n_in,
                              void* d_out, int out_size, void* d_ws, size_t ws_size,
                              hipStream_t stream) {
  const int*   src      = (const int*)d_in[0];
  const int*   speaker  = (const int*)d_in[2];
  const int*   edge_src = (const int*)d_in[4];
  const int*   edge_dst = (const int*)d_in[5];
  const int*   rels     = (const int*)d_in[6];
  const float* emb      = (const float*)d_in[7];
  const float* spk_tab  = (const float*)d_in[8];
  const float* W_ih     = (const float*)d_in[9];
  const float* W_hh     = (const float*)d_in[10];
  const float* b_lstm   = (const float*)d_in[11];
  const float* W_rel    = (const float*)d_in[12];
  const float* W_self   = (const float*)d_in[13];
  const float* b_gnn    = (const float*)d_in[14];
  const float* Wg1      = (const float*)d_in[15];
  const float* Wg2      = (const float*)d_in[16];
  float* out = (float*)d_out;

  // workspace layout (bytes)
  char* ws = (char*)d_ws;
  _Float16* X     = (_Float16*)(ws + 0);                    // 256*64*512 f16 = 16 MB
  _Float16* WT    = (_Float16*)(ws + 16777216);             // 2048*1024 f16 = 4 MB
  float* utt_h    = (float*)(ws + 20971520);                // 256*512 f32
  float* nodes    = (float*)(ws + 21495808);                // 8*33*512 f32
  float* msg      = (float*)(ws + 22036480);                // 8*128*512 f32
  float* agg      = (float*)(ws + 24133632);                // 8*33*512 f32
  float* newnodes = (float*)(ws + 24674304);                // 8*33*512 f32

  prep_x_kernel<<<Nn * Ll, 256, 0, stream>>>(src, speaker, emb, spk_tab, X);
  prep_wt_kernel<<<G4H, 256, 0, stream>>>(W_ih, W_hh, WT);
  lstm_kernel<<<Nn / 16, 512, 0, stream>>>(X, WT, b_lstm, out + OFF_MB, utt_h);
  gnn_nodes_kernel<<<(Bb * Hh) / 256, 256, 0, stream>>>(utt_h, nodes);
  gnn_msg_kernel<<<Bb * Ee, 256, 0, stream>>>(nodes, edge_src, rels, W_rel, msg);
  gnn_agg_kernel<<<Bb * (Uu + 1), 256, 0, stream>>>(msg, edge_dst, agg);
  gnn_update_kernel<<<Bb * (Uu + 1), 256, 0, stream>>>(nodes, agg, W_self, b_gnn,
                                                       Wg1, Wg2, newnodes);
  finalize_small_kernel<<<(Bb * Ss) / 256, 256, 0, stream>>>(newnodes, out);
  mbu_kernel<<<(Ss * Bb * Hh) / 256, 256, 0, stream>>>(newnodes, out + OFF_MBU);
}
